// T4_42537356100292
// MI455X (gfx1250) — matrix operand order repaired and checked by reading, not compiled, not run on hardware
//
#include <hip/hip_runtime.h>

typedef __attribute__((ext_vector_type(16))) __bf16 v16bf;
typedef __attribute__((ext_vector_type(8)))  __bf16 v8bf;
typedef __attribute__((ext_vector_type(8)))  float  v8f;
typedef __attribute__((ext_vector_type(4)))  unsigned int v4u;
typedef __attribute__((ext_vector_type(8)))  int v8i;
typedef __attribute__((ext_vector_type(4)))  int v4i;

#define B_   256
#define S_   128
#define H_   512
#define G_   2048
#define NT_  128      // 2048 / 16 N-tiles
#define TM_  16       // batch rows per block (WMMA M)
#define KT_HH  16     // 512/32 K-tiles for Whh
#define KT_IH  9      // 288/32 K-tiles for enc Wih (257 padded to 288)
#define KT_DIH 4      // 128/32 K-tiles for dec Wih (e_d part)

// LDS strides (elements) -- padded to spread banks, keep 16B alignment
#define GSTR 2064
#define HSTR 520
#define XSTR 296

#if __has_builtin(__builtin_amdgcn_tensor_load_to_lds)
#define HAVE_TDM 1
#else
#define HAVE_TDM 0
#endif

__device__ __forceinline__ float sigmoidf_(float x) {
  return 1.0f / (1.0f + __expf(-x));
}

// A fragment (16x32 bf16) per ISA layout: lane L holds row M=L&15,
// K in {half*8 .. half*8+7} U {16+half*8 .. 16+half*8+7}, half = L>>4.
__device__ __forceinline__ v16bf load_a_frag(const __bf16* base, int stride, int kt) {
  const int lane = threadIdx.x & 31;
  const int row  = lane & 15;
  const int half = lane >> 4;
  const __bf16* p = base + row * stride + kt * 32 + half * 8;
  v8bf lo = *(const v8bf*)p;
  v8bf hi = *(const v8bf*)(p + 16);
  return __builtin_shufflevector(lo, hi, 0,1,2,3,4,5,6,7,8,9,10,11,12,13,14,15);
}

// B fragment: pre-packed so each lane reads 16 contiguous bf16 (32B).
__device__ __forceinline__ v16bf load_b_frag(const __bf16* __restrict__ P, int kt, int nt) {
  const int lane = threadIdx.x & 31;
  return *(const v16bf*)(P + ((((size_t)kt * NT_ + nt) * 32 + lane) << 4));
}

__device__ __forceinline__ v8f wmma_acc(v8f acc, v16bf a, v16bf b) {
  return __builtin_amdgcn_wmma_f32_16x16x32_bf16(false, a, false, b, (short)0, acc,
                                                 false, false);
}

// TDM: DMA a 16x128 bf16 tile (rows rowStrideElems apart in memory) into LDS at
// byte offset ldsByteOff, padding each 256B row out to XSTR*2 bytes of pitch.
__device__ __forceinline__ void tdm_load_tile_16x128_bf16(const __bf16* src,
                                                          unsigned ldsByteOff,
                                                          unsigned rowStrideElems,
                                                          unsigned padDwMinus1) {
#if HAVE_TDM
  unsigned long long ga = (unsigned long long)(uintptr_t)src;
  v4u g0;
  g0[0] = 1u;                                    // count=1 (valid), user mode
  g0[1] = ldsByteOff;                            // lds_addr (bytes)
  g0[2] = (unsigned)(ga & 0xffffffffu);          // global_addr[31:0]
  g0[3] = (unsigned)((ga >> 32) & 0x01ffffffu)   // global_addr[56:32]
          | (2u << 30);                          // type = 2 ("image")
  v8i g1;
  g1[0] = (int)((1u << 16)                       // data_size = 2 bytes (bf16)
              | (1u << 20)                       // pad_enable (LDS row pitch)
              | (5u << 22)                       // pad_interval: 64 DWORDs (256B)
              | (padDwMinus1 << 25));            // pad_amount: (code+1) DWORDs
  g1[1] = (int)(128u << 16);                     // tensor_dim0 = 128 elements
  g1[2] = (int)((unsigned)TM_ << 16);            // tensor_dim1 = 16 rows
  g1[3] = (int)(128u << 16);                     // tile_dim0 = 128
  g1[4] = TM_;                                   // tile_dim1 = 16 rows
  g1[5] = (int)rowStrideElems;                   // tensor_dim0_stride (elements)
  g1[6] = 0;
  g1[7] = 0;
  v4i z4 = {};
#if defined(__clang_major__) && (__clang_major__ >= 23)
  v8i z8 = {};
  __builtin_amdgcn_tensor_load_to_lds(g0, g1, z4, z4, z8, 0);
#else
  __builtin_amdgcn_tensor_load_to_lds(g0, g1, z4, z4, 0);
#endif
  __builtin_amdgcn_s_wait_tensorcnt(0);
#else
  (void)src; (void)ldsByteOff; (void)rowStrideElems; (void)padDwMinus1;
#endif
}

// K rolled (no cross-iteration hoisting -> no spills); within one kt iteration
// the 8 B-fragment pairs are issued as a batch ahead of the 8-WMMA chain so the
// scheduler emits graduated loadcnt waits instead of a full drain per WMMA.
template <int NKT>
__device__ __forceinline__ void gemm_panel(const __bf16* aBase, int aStride,
                                           const __bf16* __restrict__ P, int ntBase,
                                           v8f* acc) {
#pragma unroll 1
  for (int kt = 0; kt < NKT; ++kt) {
    v16bf a  = load_a_frag(aBase, aStride, kt);
    v16bf b0 = load_b_frag(P, kt, ntBase + 0);
    v16bf b1 = load_b_frag(P, kt, ntBase + 1);
    v16bf b2 = load_b_frag(P, kt, ntBase + 2);
    v16bf b3 = load_b_frag(P, kt, ntBase + 3);
    v16bf b4 = load_b_frag(P, kt, ntBase + 4);
    v16bf b5 = load_b_frag(P, kt, ntBase + 5);
    v16bf b6 = load_b_frag(P, kt, ntBase + 6);
    v16bf b7 = load_b_frag(P, kt, ntBase + 7);
    acc[0] = wmma_acc(acc[0], a, b0);
    acc[1] = wmma_acc(acc[1], a, b1);
    acc[2] = wmma_acc(acc[2], a, b2);
    acc[3] = wmma_acc(acc[3], a, b3);
    acc[4] = wmma_acc(acc[4], a, b4);
    acc[5] = wmma_acc(acc[5], a, b5);
    acc[6] = wmma_acc(acc[6], a, b6);
    acc[7] = wmma_acc(acc[7], a, b7);
  }
}

__device__ __forceinline__ void store_g(float* gL, const v8f* acc, int ntBase) {
  const int lane = threadIdx.x & 31;
  const int mb   = (lane >> 4) * 8;
#pragma unroll
  for (int i = 0; i < 8; ++i) {
    const int col = (ntBase + i) * 16 + (lane & 15);
#pragma unroll
    for (int r = 0; r < 8; ++r) gL[(mb + r) * GSTR + col] = acc[i][r];
  }
}

// ---------------------------------------------------------------------------
// Pack fp32 weight (K x N row-major) into WMMA-B bf16 fragment layout.
// ---------------------------------------------------------------------------
__global__ __launch_bounds__(256) void pack_w_kernel(const float* __restrict__ W,
                                                     __bf16* __restrict__ P,
                                                     int K, int KT, int N) {
  const int NT = N >> 4;
  size_t i = (size_t)blockIdx.x * blockDim.x + threadIdx.x;
  size_t total = (size_t)KT * 32 * N;
  if (i >= total) return;
  int e  = (int)(i & 15);
  int l  = (int)((i >> 4) & 31);
  size_t tile = i >> 9;
  int nt = (int)(tile % NT);
  int kt = (int)(tile / NT);
  int k = kt * 32 + (l >> 4) * 8 + (e & 7) + ((e >> 3) << 4);
  int n = nt * 16 + (l & 15);
  P[i] = (k < K) ? (__bf16)W[(size_t)k * N + n] : (__bf16)0.0f;
}

// e_x = relu(temporal_x @ temp_W + temp_b)  -> bf16
__global__ __launch_bounds__(256) void embed_temporal_kernel(const float* __restrict__ tx,
                                                             const float* __restrict__ W,
                                                             const float* __restrict__ b,
                                                             __bf16* __restrict__ eX) {
  size_t i = (size_t)blockIdx.x * 256 + threadIdx.x; // rs*128 + j
  int j = (int)(i & 127);
  size_t rs = i >> 7;
  const float* x = tx + rs * 64;
  float acc = b[j];
#pragma unroll 8
  for (int k = 0; k < 64; ++k) acc += x[k] * W[k * 128 + j];
  eX[i] = (__bf16)fmaxf(acc, 0.0f);
}

// e_d = relu(static_x @ stat_W + stat_b) -> bf16
__global__ __launch_bounds__(256) void embed_static_kernel(const float* __restrict__ sx,
                                                           const float* __restrict__ W,
                                                           const float* __restrict__ b,
                                                           __bf16* __restrict__ eD) {
  int i = blockIdx.x * 256 + threadIdx.x; // b*128 + j
  int j = i & 127;
  int bb = i >> 7;
  const float* x = sx + bb * 32;
  float acc = b[j];
#pragma unroll 8
  for (int k = 0; k < 32; ++k) acc += x[k] * W[k * 128 + j];
  eD[i] = (__bf16)fmaxf(acc, 0.0f);
}

// ---------------------------------------------------------------------------
// Encoder LSTM: one block per 16 batch rows, 8 waves, 128 serial steps.
// ---------------------------------------------------------------------------
__global__ __launch_bounds__(256) void encoder_kernel(const __bf16* __restrict__ eX,
                                                      const __bf16* __restrict__ eD,
                                                      const float*  __restrict__ hist,
                                                      const __bf16* __restrict__ Pih,
                                                      const __bf16* __restrict__ Phh,
                                                      const float*  __restrict__ bias,
                                                      float* __restrict__ enc_out,
                                                      float* __restrict__ h_n,
                                                      float* __restrict__ c_n) {
  extern __shared__ char smem[];
  float*  gL = (float*)smem;                                    // TM_ x GSTR fp32
  __bf16* hL = (__bf16*)(smem + (size_t)TM_ * GSTR * 4);        // TM_ x HSTR bf16
  __bf16* xL = (__bf16*)(smem + (size_t)TM_ * GSTR * 4 + (size_t)TM_ * HSTR * 2);
  float*  bL = (float*)(smem + (size_t)TM_ * GSTR * 4 + (size_t)TM_ * HSTR * 2 +
                        (size_t)TM_ * XSTR * 2);                // 2048 fp32 bias
  float*  cL = bL + G_;                                         // TM_ x H_ cell state

  const int tid  = threadIdx.x;
  const int b0   = blockIdx.x * TM_;
  const int wave = tid >> 5;

  // h = 0 ; c = 0 ; bias -> LDS
  for (int i = tid; i < TM_ * HSTR; i += 256) hL[i] = (__bf16)0.0f;
  for (int i = tid; i < TM_ * H_; i += 256) cL[i] = 0.0f;
  for (int i = tid; i < G_; i += 256) bL[i] = bias[i];
  // x_t columns 128..255 = e_d (constant across steps)
  for (int i = tid; i < TM_ * 128; i += 256) {
    int row = i >> 7, k = i & 127;
    xL[row * XSTR + 128 + k] = eD[(b0 + row) * 128 + k];
  }
  // zero K-pad region (cols 256..XSTR-1); col 256 rewritten per step
  for (int i = tid; i < TM_ * (XSTR - 256); i += 256) {
    int row = i / (XSTR - 256), k = i % (XSTR - 256);
    xL[row * XSTR + 256 + k] = (__bf16)0.0f;
  }

  for (int s = 0; s < S_; ++s) {
    // stream this step's embedded inputs into LDS (TDM DMA when available)
#if HAVE_TDM
    if (tid < 32) {
      // pad each 256B row out to XSTR*2 = 592B pitch: 336B = 84 DWORDs -> code 83
      tdm_load_tile_16x128_bf16(&eX[(((size_t)b0) * S_ + s) * 128],
                                (unsigned)((size_t)TM_ * GSTR * 4 +
                                           (size_t)TM_ * HSTR * 2),
                                (unsigned)(S_ * 128), 83u);
    }
#else
    for (int i = tid; i < TM_ * 128; i += 256) {
      int row = i >> 7, k = i & 127;
      xL[row * XSTR + k] = eX[(((size_t)(b0 + row)) * S_ + s) * 128 + k];
    }
#endif
    if (tid < TM_) xL[tid * XSTR + 256] = (__bf16)hist[(b0 + tid) * S_ + s];
    if (s + 1 < S_)
      __builtin_prefetch(&eX[(((size_t)(b0 + (tid >> 4))) * S_ + s + 1) * 128 +
                             ((tid & 15) << 3)], 0, 1);
    __syncthreads();

    // g = x_t @ Wih + h @ Whh : per wave 16 N-tiles, 2 passes of 8 accumulators
#pragma unroll 1
    for (int pass = 0; pass < 2; ++pass) {
      const int ntBase = wave * 16 + pass * 8;
      v8f acc[8] = {};
      gemm_panel<KT_IH>(xL, XSTR, Pih, ntBase, acc);
      gemm_panel<KT_HH>(hL, HSTR, Phh, ntBase, acc);
      store_g(gL, acc, ntBase);
    }
    __syncthreads();

    // gates: thread -> 32 contiguous (row,u) cells; cell state in LDS
#pragma unroll 4
    for (int j = 0; j < 32; ++j) {
      int idx = tid * 32 + j;
      int row = idx >> 9, u = idx & 511;
      float gi = gL[row * GSTR + u]         + bL[u];
      float gf = gL[row * GSTR + 512 + u]   + bL[512 + u];
      float gg = gL[row * GSTR + 1024 + u]  + bL[1024 + u];
      float go = gL[row * GSTR + 1536 + u]  + bL[1536 + u];
      float cn = sigmoidf_(gf) * cL[idx] + sigmoidf_(gi) * tanhf(gg);
      float hn = sigmoidf_(go) * tanhf(cn);
      cL[idx] = cn;
      hL[row * HSTR + u] = (__bf16)hn;
      enc_out[(((size_t)(b0 + row)) * S_ + s) * H_ + u] = hn;
      if (s == S_ - 1) {
        h_n[(size_t)(b0 + row) * H_ + u] = hn;
        c_n[(size_t)(b0 + row) * H_ + u] = cn;
      }
    }
    __syncthreads();
  }
}

// propensity logits + precomputed attention term enc_We = enc_out . We
__global__ __launch_bounds__(256) void prop_attn_kernel(const float* __restrict__ enc_out,
                                                        const float* __restrict__ prop_W,
                                                        const float* __restrict__ prop_b,
                                                        const float* __restrict__ attn_W,
                                                        float* __restrict__ out_prop,
                                                        float* __restrict__ enc_We) {
  int i = blockIdx.x * 256 + threadIdx.x; // b*S + s
  const float* ep = enc_out + (size_t)i * H_;
  float p = 0.f, we = 0.f;
#pragma unroll 8
  for (int h = 0; h < H_; ++h) {
    float v = ep[h];
    p += v * prop_W[h];
    we += v * attn_W[H_ + h];
  }
  out_prop[i] = p + prop_b[0];
  enc_We[i] = we;
}

// ---------------------------------------------------------------------------
// Decoder LSTM + attention.
// ---------------------------------------------------------------------------
__global__ __launch_bounds__(256) void decoder_kernel(const __bf16* __restrict__ eD,
                                                      const float*  __restrict__ fut_tr,
                                                      const float*  __restrict__ fut_init,
                                                      const __bf16* __restrict__ PdWih,
                                                      const __bf16* __restrict__ PdWhh,
                                                      const float*  __restrict__ dWih,
                                                      const float*  __restrict__ dbias,
                                                      const float*  __restrict__ attn_W,
                                                      const float*  __restrict__ attn_b,
                                                      const float*  __restrict__ out_W,
                                                      const float*  __restrict__ out_b,
                                                      const float*  __restrict__ enc_out,
                                                      const float*  __restrict__ enc_We,
                                                      float* __restrict__ gbase,
                                                      const float* __restrict__ h_n,
                                                      const float* __restrict__ c_n,
                                                      float* __restrict__ preds) {
  extern __shared__ char smem[];
  float*  gL = (float*)smem;                                        // TM_ x GSTR
  __bf16* hL = (__bf16*)(smem + (size_t)TM_ * GSTR * 4);            // TM_ x HSTR
  __bf16* dL = (__bf16*)(smem + (size_t)TM_ * GSTR * 4 + (size_t)TM_ * HSTR * 2); // 16x128
  float*  hF = (float*)(smem + (size_t)TM_ * GSTR * 4 + (size_t)TM_ * HSTR * 2 +
                        (size_t)TM_ * 128 * 2);                      // TM_ x H_ fp32
  float*  wyL = (float*)((char*)hF + (size_t)TM_ * H_ * 4);          // dec_Wih row 128
  float*  waL = wyL + G_;                                            // dec_Wih row 129
  float*  cL  = waL + G_;                                            // TM_ x H_ cell
  float*  sm  = cL + TM_ * H_;
  float* hWh = sm;       // 16
  float* yL  = sm + 16;  // 16
  float* aL  = sm + 32;  // 16
  // overlays inside the (step-dead) g region:
  float* hWi = gL;                       // 16x512
  float* scr = gL + TM_ * H_;            // 16x128 (scores, then softmax weights)
  float* ctx = gL + TM_ * H_ + TM_ * S_; // 16x512

  const int tid  = threadIdx.x;
  const int b0   = blockIdx.x * TM_;
  const int wave = tid >> 5;

  // init: e_d tile, rank-1 weight rows, h/c from encoder, y from init outcome
  for (int i = tid; i < TM_ * 128; i += 256) {
    int row = i >> 7, k = i & 127;
    dL[row * 128 + k] = eD[(b0 + row) * 128 + k];
  }
  for (int i = tid; i < G_; i += 256) {
    wyL[i] = dWih[(size_t)128 * G_ + i];
    waL[i] = dWih[(size_t)129 * G_ + i];
  }
  for (int j = 0; j < 32; ++j) {
    int idx = tid * 32 + j;
    int row = idx >> 9, u = idx & 511;
    cL[idx] = c_n[(size_t)(b0 + row) * H_ + u];
    hL[row * HSTR + u] = (__bf16)h_n[(size_t)(b0 + row) * H_ + u];
  }
  if (tid < TM_) yL[tid] = fut_init[b0 + tid];
  __syncthreads();

  // gbase = e_d @ dec_Wih[0:128,:] + dec_b  (written to global, read each step)
#pragma unroll 1
  for (int pass = 0; pass < 2; ++pass) {
    const int ntBase = wave * 16 + pass * 8;
    v8f acc[8] = {};
    gemm_panel<KT_DIH>(dL, 128, PdWih, ntBase, acc);
    const int lane = tid & 31;
    const int mb   = (lane >> 4) * 8;
#pragma unroll
    for (int i = 0; i < 8; ++i) {
      const int col = (ntBase + i) * 16 + (lane & 15);
#pragma unroll
      for (int r = 0; r < 8; ++r)
        gbase[((size_t)(b0 + mb + r)) * G_ + col] = acc[i][r] + dbias[col];
    }
  }
  __threadfence();
  __syncthreads();

  for (int t = 0; t < 16; ++t) {
    if (tid < TM_) aL[tid] = fut_tr[(b0 + tid) * 16 + t];
    __syncthreads();

    // g (recurrent part) = h @ dec_Whh
#pragma unroll 1
    for (int pass = 0; pass < 2; ++pass) {
      const int ntBase = wave * 16 + pass * 8;
      v8f acc[8] = {};
      gemm_panel<KT_HH>(hL, HSTR, PdWhh, ntBase, acc);
      store_g(gL, acc, ntBase);
    }
    __syncthreads();

    // gates (+ gbase + rank-1 y/a input terms); cell state in LDS
#pragma unroll 4
    for (int j = 0; j < 32; ++j) {
      int idx = tid * 32 + j;
      int row = idx >> 9, u = idx & 511;
      float yv = yL[row], av = aL[row];
      const float* gb = gbase + ((size_t)(b0 + row)) * G_;
      float gi = gL[row * GSTR + u]        + gb[u]        + yv * wyL[u]        + av * waL[u];
      float gf = gL[row * GSTR + 512 + u]  + gb[512 + u]  + yv * wyL[512 + u]  + av * waL[512 + u];
      float gg = gL[row * GSTR + 1024 + u] + gb[1024 + u] + yv * wyL[1024 + u] + av * waL[1024 + u];
      float go = gL[row * GSTR + 1536 + u] + gb[1536 + u] + yv * wyL[1536 + u] + av * waL[1536 + u];
      float cn = sigmoidf_(gf) * cL[idx] + sigmoidf_(gi) * tanhf(gg);
      float hn = sigmoidf_(go) * tanhf(cn);
      cL[idx] = cn;
      hL[row * HSTR + u] = (__bf16)hn;
      hF[row * H_ + u] = hn;
    }
    __syncthreads();

    // attention precompute: hWi = h * Wi ; hWh = h . Wh
    for (int j = 0; j < 32; ++j) {
      int idx = tid * 32 + j;
      int row = idx >> 9, u = idx & 511;
      hWi[row * H_ + u] = hF[row * H_ + u] * attn_W[2 * H_ + u];
    }
    if (tid < TM_) {
      float a = 0.f;
      for (int h = 0; h < H_; ++h) a += hF[tid * H_ + h] * attn_W[h];
      hWh[tid] = a;
    }
    __syncthreads();

    // scores[row][s] = tanh(hWh + enc_We + enc_out . hWi + b)
    for (int e = tid * 8; e < tid * 8 + 8; ++e) {
      int row = e >> 7, s = e & 127;
      const float* ep = enc_out + (((size_t)(b0 + row)) * S_ + s) * H_;
      float dot = 0.f;
#pragma unroll 8
      for (int h = 0; h < H_; ++h) dot += ep[h] * hWi[row * H_ + h];
      scr[row * S_ + s] = tanhf(hWh[row] + enc_We[(b0 + row) * S_ + s] + dot + attn_b[0]);
    }
    __syncthreads();

    // softmax over s per row
    if (tid < TM_) {
      float m = -1e30f;
      for (int s = 0; s < S_; ++s) m = fmaxf(m, scr[tid * S_ + s]);
      float sum = 0.f;
      for (int s = 0; s < S_; ++s) {
        float ev = __expf(scr[tid * S_ + s] - m);
        scr[tid * S_ + s] = ev;
        sum += ev;
      }
      float inv = 1.0f / sum;
      for (int s = 0; s < S_; ++s) scr[tid * S_ + s] *= inv;
    }
    __syncthreads();

    // context[row][u] = sum_s w * enc_out
    for (int j = 0; j < 32; ++j) {
      int idx = tid * 32 + j;
      int row = idx >> 9, u = idx & 511;
      float a = 0.f;
      for (int s = 0; s < S_; ++s)
        a += scr[row * S_ + s] * enc_out[(((size_t)(b0 + row)) * S_ + s) * H_ + u];
      ctx[row * H_ + u] = a;
    }
    __syncthreads();

    // pred = [h, context, a] @ out_W + out_b ; feeds y for next step
    if (tid < TM_) {
      float a = out_b[0] + aL[tid] * out_W[2 * H_];
      for (int h = 0; h < H_; ++h) a += hF[tid * H_ + h] * out_W[h];
      for (int h = 0; h < H_; ++h) a += ctx[tid * H_ + h] * out_W[H_ + h];
      preds[(b0 + tid) * 16 + t] = a;
      yL[tid] = a;
    }
    __syncthreads();
  }
}

// ---------------------------------------------------------------------------
extern "C" void kernel_launch(void* const* d_in, const int* in_sizes, int n_in,
                              void* d_out, int out_size, void* d_ws, size_t ws_size,
                              hipStream_t stream) {
  const float* temporal_x = (const float*)d_in[0];
  const float* static_x   = (const float*)d_in[1];
  const float* hist_tr    = (const float*)d_in[2];
  const float* fut_tr     = (const float*)d_in[3];
  const float* fut_init   = (const float*)d_in[4];
  const float* temp_W = (const float*)d_in[5];
  const float* temp_b = (const float*)d_in[6];
  const float* stat_W = (const float*)d_in[7];
  const float* stat_b = (const float*)d_in[8];
  const float* enc_Wih = (const float*)d_in[9];
  const float* enc_Whh = (const float*)d_in[10];
  const float* enc_b   = (const float*)d_in[11];
  const float* attn_W  = (const float*)d_in[12];
  const float* attn_b  = (const float*)d_in[13];
  const float* dec_Wih = (const float*)d_in[14];
  const float* dec_Whh = (const float*)d_in[15];
  const float* dec_b   = (const float*)d_in[16];
  const float* out_W   = (const float*)d_in[17];
  const float* out_b   = (const float*)d_in[18];
  const float* prop_W  = (const float*)d_in[19];
  const float* prop_b  = (const float*)d_in[20];

  float* out_preds = (float*)d_out;            // (256,16,1)
  float* out_prop  = (float*)d_out + 256 * 16; // (256,128,1)

  char* w = (char*)d_ws;
  size_t off = 0;
  auto alloc = [&](size_t bytes) {
    char* p = w + off;
    off += (bytes + 255) & ~(size_t)255;
    return p;
  };
  __bf16* P_eWih = (__bf16*)alloc((size_t)KT_IH * 32 * G_ * 2);
  __bf16* P_eWhh = (__bf16*)alloc((size_t)KT_HH * 32 * G_ * 2);
  __bf16* P_dWih = (__bf16*)alloc((size_t)KT_DIH * 32 * G_ * 2);
  __bf16* P_dWhh = (__bf16*)alloc((size_t)KT_HH * 32 * G_ * 2);
  __bf16* eX     = (__bf16*)alloc((size_t)B_ * S_ * 128 * 2);
  __bf16* eDb    = (__bf16*)alloc((size_t)B_ * 128 * 2);
  float* enc_out = (float*)alloc((size_t)B_ * S_ * H_ * 4);
  float* enc_Wep = (float*)alloc((size_t)B_ * S_ * 4);
  float* h_n     = (float*)alloc((size_t)B_ * H_ * 4);
  float* c_n     = (float*)alloc((size_t)B_ * H_ * 4);
  float* gbase   = (float*)alloc((size_t)B_ * G_ * 4);

  // one-time weight packing (bf16, WMMA B-fragment layout)
  pack_w_kernel<<<(KT_IH * 32 * G_ + 255) / 256, 256, 0, stream>>>(enc_Wih, P_eWih, 257, KT_IH, G_);
  pack_w_kernel<<<(KT_HH * 32 * G_ + 255) / 256, 256, 0, stream>>>(enc_Whh, P_eWhh, 512, KT_HH, G_);
  pack_w_kernel<<<(KT_DIH * 32 * G_ + 255) / 256, 256, 0, stream>>>(dec_Wih, P_dWih, 128, KT_DIH, G_);
  pack_w_kernel<<<(KT_HH * 32 * G_ + 255) / 256, 256, 0, stream>>>(dec_Whh, P_dWhh, 512, KT_HH, G_);

  embed_temporal_kernel<<<(B_ * S_ * 128) / 256, 256, 0, stream>>>(temporal_x, temp_W, temp_b, eX);
  embed_static_kernel<<<(B_ * 128) / 256, 256, 0, stream>>>(static_x, stat_W, stat_b, eDb);

  const size_t encL = (size_t)TM_ * GSTR * 4 + (size_t)TM_ * HSTR * 2 +
                      (size_t)TM_ * XSTR * 2 + (size_t)G_ * 4 + (size_t)TM_ * H_ * 4;
  encoder_kernel<<<B_ / TM_, 256, encL, stream>>>(eX, eDb, hist_tr, P_eWih, P_eWhh, enc_b,
                                                  enc_out, h_n, c_n);

  prop_attn_kernel<<<(B_ * S_) / 256, 256, 0, stream>>>(enc_out, prop_W, prop_b, attn_W,
                                                        out_prop, enc_Wep);

  const size_t decL = (size_t)TM_ * GSTR * 4 + (size_t)TM_ * HSTR * 2 +
                      (size_t)TM_ * 128 * 2 + (size_t)TM_ * H_ * 4 +
                      (size_t)2 * G_ * 4 + (size_t)TM_ * H_ * 4 + 192;
  decoder_kernel<<<B_ / TM_, 256, decL, stream>>>(eDb, fut_tr, fut_init, P_dWih, P_dWhh,
                                                  dec_Wih, dec_b, attn_W, attn_b, out_W, out_b,
                                                  enc_out, enc_Wep, gbase, h_n, c_n, out_preds);

  (void)in_sizes; (void)n_in; (void)out_size; (void)ws_size;
}
